// VoxelBackbone_79164837199887
// MI455X (gfx1250) — matrix operand order repaired and checked by reading, not compiled, not run on hardware
//
#include <hip/hip_runtime.h>
#include <hip/hip_bf16.h>
#include <stdint.h>

typedef _Float16 f16;
typedef __attribute__((ext_vector_type(8)))  f16   v8h;
typedef __attribute__((ext_vector_type(16))) f16   v16h;
typedef __attribute__((ext_vector_type(8)))  float v8f;
typedef __attribute__((ext_vector_type(4)))  int   v4i;

#define DEVINL static __device__ __forceinline__

// ---------------------------------------------------------------------------
// CDNA5 async global->LDS copy (ASYNCcnt) -- guarded so compile never breaks
// ROCm 7.2 signature (from probe): (v4i* src, v4i* dst, imm offset, imm cpol)
// with generic pointers; compiler infers the address spaces.
// ---------------------------------------------------------------------------
#if defined(__HIP_DEVICE_COMPILE__) && __has_builtin(__builtin_amdgcn_global_load_async_to_lds_b128)
#define HAVE_ASYNC_LDS 1
#else
#define HAVE_ASYNC_LDS 0
#endif

#if HAVE_ASYNC_LDS
DEVINL void asy_b128(const void* g, void* l) {
    __builtin_amdgcn_global_load_async_to_lds_b128((v4i*)g, (v4i*)l, 0, 0);
}
template <int N> DEVINL void asy_wait() {
#if __has_builtin(__builtin_amdgcn_s_wait_asynccnt)
    __builtin_amdgcn_s_wait_asynccnt(N);
#else
    asm volatile("s_wait_asynccnt %0" :: "i"(N) : "memory");
#endif
}
// stage one 16x128-f16 tile (4 KB) with 8 async b128 issues (32 lanes x 16B)
DEVINL void stage_tile_async(const f16* g, f16* l, int lane) {
#pragma unroll
    for (int it = 0; it < 8; ++it) {
        const int c = it * 32 + lane;   // 16-byte chunk id, 256 total
        asy_b128((const char*)g + c * 16, (char*)l + c * 16);
    }
}
#endif

// ---------------------------------------------------------------------------
// WMMA helpers (CDNA5 wave32, D = A(16x32 f16) x B(32x16 f16) + C(16x16 f32))
// ---------------------------------------------------------------------------
DEVINL v8f wmma32(v16h a, v16h b, v8f c) {
    return __builtin_amdgcn_wmma_f32_16x16x32_f16(
        /*neg_a=*/false, a, /*neg_b=*/false, b,
        /*c_mod=*/(short)0, c, /*reuse_a=*/false, /*reuse_b=*/false);
}

// A fragment (16x32, 16-bit): lane(0-15)=row M, holds K=0..7 & 16..23;
// lane(16-31)=row, holds K=8..15 & 24..31.  base = (row0,k0); ld in elements.
DEVINL v16h ldA(const f16* base, int ld, int lane) {
    const f16* p = base + (size_t)(lane & 15) * ld + ((lane < 16) ? 0 : 8);
    v8h lo = *(const v8h*)(p);
    v8h hi = *(const v8h*)(p + 16);
    return __builtin_shufflevector(lo, hi, 0,1,2,3,4,5,6,7,8,9,10,11,12,13,14,15);
}

// B fragment (32x16, 16-bit), NT form: memory holds the N dimension as rows
// of length K.  lane(0-15)=col N holds K=0..7 & 16..23; lane(16-31) holds K=8..15 & 24..31.
DEVINL v16h ldB_nt(const f16* base, int ld, int lane) {
    const f16* p = base + (size_t)(lane & 15) * ld + ((lane < 16) ? 0 : 8);
    v8h lo = *(const v8h*)(p);
    v8h hi = *(const v8h*)(p + 16);
    return __builtin_shufflevector(lo, hi, 0,1,2,3,4,5,6,7,8,9,10,11,12,13,14,15);
}

static constexpr float RSBN = 0.9999950000374998f;  // 1/sqrt(1+1e-5)

// ---------------------------------------------------------------------------
// Kernel 1: fused VFE1 -> VFE2 -> FCN -> max over T -> LayerNorm   (per b,k)
// ---------------------------------------------------------------------------
__global__ __launch_bounds__(128) void vfe_fused(
    const float* __restrict__ vox,
    const float* __restrict__ W1, const float* __restrict__ b1,
    const float* __restrict__ g1, const float* __restrict__ be1,
    const float* __restrict__ W2, const float* __restrict__ b2,
    const float* __restrict__ g2, const float* __restrict__ be2,
    const float* __restrict__ W3, const float* __restrict__ b3,
    const float* __restrict__ g3, const float* __restrict__ be3,
    const float* __restrict__ lng, const float* __restrict__ lnb,
    f16* __restrict__ Xout)
{
    const int bkid = blockIdx.x;
    const int b = bkid >> 12, k = bkid & 4095;
    const int tid = threadIdx.x;
    const int rowbase = k * 35;               // row index within batch = k*T+t

    __shared__ float s_in[35][8];
    __shared__ float s_mask[35];
    __shared__ float s_a[35][32];
    __shared__ float s_b[35][128];
    __shared__ float s_c[35][128];
    __shared__ float s_laf[64];
    __shared__ float s_red[128];

    for (int idx = tid; idx < 35 * 7; idx += 128) {
        int t = idx / 7, f = idx % 7;
        s_in[t][f] = vox[(((size_t)(b * 4096 + k) * 35 + t) * 7 + f)];
    }
    __syncthreads();
    if (tid < 35) {
        float mx = s_in[tid][0];
#pragma unroll
        for (int f = 1; f < 7; ++f) mx = fmaxf(mx, s_in[tid][f]);
        s_mask[tid] = (mx != 0.0f) ? 1.0f : 0.0f;
    }
    __syncthreads();

    // ---- VFE1 linear 7->16 + "viewed" BN + relu ----
    for (int idx = tid; idx < 35 * 16; idx += 128) {
        int t = idx >> 4, f = idx & 15;
        float acc = b1[f];
#pragma unroll
        for (int c = 0; c < 7; ++c) acc += s_in[t][c] * W1[c * 16 + f];
        int ch = ((rowbase + t) * 16 + f) / 143360;   // K*T = 143360
        float y = acc * (g1[ch] * RSBN) + be1[ch];
        s_a[t][f] = fmaxf(y, 0.0f);
    }
    __syncthreads();
    if (tid < 16) {
        float mx = -1e30f;
        for (int t = 0; t < 35; ++t) mx = fmaxf(mx, s_a[t][tid]);
        s_laf[tid] = mx;
    }
    __syncthreads();
    for (int idx = tid; idx < 35 * 32; idx += 128) {
        int t = idx >> 5, f = idx & 31;
        float v = (f < 16) ? s_a[t][f] : s_laf[f - 16];
        s_a[t][f] = v * s_mask[t];
    }
    __syncthreads();

    // ---- VFE2 linear 32->64 ----
    for (int idx = tid; idx < 35 * 64; idx += 128) {
        int t = idx >> 6, f = idx & 63;
        float acc = b2[f];
#pragma unroll 8
        for (int c = 0; c < 32; ++c) acc += s_a[t][c] * W2[c * 64 + f];
        int ch = ((rowbase + t) * 64 + f) / 143360;
        float y = acc * (g2[ch] * RSBN) + be2[ch];
        s_c[t][f] = fmaxf(y, 0.0f);
    }
    __syncthreads();
    if (tid < 64) {
        float mx = -1e30f;
        for (int t = 0; t < 35; ++t) mx = fmaxf(mx, s_c[t][tid]);
        s_laf[tid] = mx;
    }
    __syncthreads();
    for (int idx = tid; idx < 35 * 128; idx += 128) {
        int t = idx >> 7, f = idx & 127;
        float v = (f < 64) ? s_c[t][f] : s_laf[f - 64];
        s_b[t][f] = v * s_mask[t];
    }
    __syncthreads();

    // ---- FCN 128->128, one feature per thread ----
    {
        const int f = tid;
        for (int t = 0; t < 35; ++t) {
            float acc = b3[f];
#pragma unroll 8
            for (int c = 0; c < 128; ++c) acc += s_b[t][c] * W3[c * 128 + f];
            int ch = ((rowbase + t) * 128 + f) / 143360;
            s_c[t][f] = fmaxf(acc * (g3[ch] * RSBN) + be3[ch], 0.0f);
        }
        __syncthreads();
        float mx = -1e30f;
        for (int t = 0; t < 35; ++t) mx = fmaxf(mx, s_c[t][f]);
        // LayerNorm over 128
        s_red[f] = mx; __syncthreads();
        for (int s = 64; s > 0; s >>= 1) { if (f < s) s_red[f] += s_red[f + s]; __syncthreads(); }
        float mean = s_red[0] * (1.0f / 128.0f);
        __syncthreads();
        float d = mx - mean;
        s_red[f] = d * d; __syncthreads();
        for (int s = 64; s > 0; s >>= 1) { if (f < s) s_red[f] += s_red[f + s]; __syncthreads(); }
        float var = s_red[0] * (1.0f / 128.0f);
        float o = d * rsqrtf(var + 1e-6f) * lng[f] + lnb[f];
        Xout[(size_t)(b * 4096 + k) * 128 + f] = (f16)o;
    }
}

// ---------------------------------------------------------------------------
// Kernel 2: NT GEMM, M x 128 x 128: out[i,o] = A[i,:].W[o,:] + bias (+res)
// one 16-row i-tile per wave; compile-time residual / transposed-store.
// ---------------------------------------------------------------------------
template <bool RES, bool TSTORE>
__global__ __launch_bounds__(128) void gemm_nt_128(
    const f16* __restrict__ A, const f16* __restrict__ W,
    const float* __restrict__ bias, const f16* __restrict__ residual,
    f16* __restrict__ out, int M)
{
    const int lane = threadIdx.x & 31;
    const int wave = threadIdx.x >> 5;
    const int i0 = blockIdx.x * 64 + wave * 16;
    const int alo = (lane < 16);

    v16h a[4];
#pragma unroll
    for (int kk = 0; kk < 4; ++kk)
        a[kk] = ldA(A + (size_t)i0 * 128 + kk * 32, 128, lane);

#pragma unroll
    for (int nt = 0; nt < 8; ++nt) {
        v8f acc = {};
#pragma unroll
        for (int kk = 0; kk < 4; ++kk) {
            v16h bf = ldB_nt(W + (size_t)(nt * 16) * 128 + kk * 32, 128, lane);
            acc = wmma32(a[kk], bf, acc);
        }
        const int co = nt * 16 + (lane & 15);
        const float bb = bias[co];
#pragma unroll
        for (int v = 0; v < 8; ++v) {
            int m = i0 + v + (alo ? 0 : 8);
            float val = acc[v] + bb;
            if constexpr (RES) val += (float)residual[(size_t)m * 128 + co];
            if constexpr (TSTORE) out[(size_t)co * M + m] = (f16)val;
            else                  out[(size_t)m * 128 + co] = (f16)val;
        }
    }
}

// ---------------------------------------------------------------------------
// Kernel 3: attention core for one 16-row i-tile (per batch).
//   phase1: S[16,4096] = (Qt_i . Kt^T)/sqrt(128) via WMMA -> LDS (f16),
//           Kt tiles double-buffered into LDS with async DMA (ASYNCcnt)
//   phase2: softmax rows in LDS (denominator folded into epilogue)
//   phase3: Ot[i,c] = sum_j P[i,j] * V2[c,j]  via WMMA (A from LDS, B NT)
// Dynamic LDS: 16*4096 f16 + stats + 4 waves * 2 * 4KB staging  (~161 KB)
// ---------------------------------------------------------------------------
__global__ __launch_bounds__(128) void attn_core(
    const f16* __restrict__ Qt, const f16* __restrict__ Kt,
    const f16* __restrict__ V2, f16* __restrict__ Ot)
{
    extern __shared__ float smem[];
    f16*   Sh     = (f16*)smem;                       // 16*4096 f16 (128 KB)
    float* s_part = (float*)(Sh + 16 * 4096);         // 128
    float* s_max  = s_part + 128;                     // 16
    float* s_inv  = s_max + 16;                       // 16
    f16*   s_stg  = (f16*)(s_inv + 16);               // 4 waves * 2 * 2048 f16

    const int tid  = threadIdx.x;
    const int lane = tid & 31;
    const int wave = tid >> 5;
    const int alo  = (lane < 16);
    const int i0   = blockIdx.x * 16;
    const float SCALE = 0.08838834764831845f;   // 1/sqrt(128)

    // ---- phase 1: scores ----
    v16h qa[4];
#pragma unroll
    for (int kk = 0; kk < 4; ++kk)
        qa[kk] = ldA(Qt + (size_t)i0 * 128 + kk * 32, 128, lane);

#if HAVE_ASYNC_LDS
    {
        f16* stg = s_stg + (size_t)wave * 2 * 2048;   // two 16x128 buffers
        stage_tile_async(Kt + (size_t)(wave * 64) * 16 * 128, stg, lane);
        for (int idx = 0; idx < 64; ++idx) {
            const int jt = wave * 64 + idx;
            const int j0 = jt * 16;
            f16* curb = stg + (idx & 1) * 2048;
            f16* nxtb = stg + ((idx + 1) & 1) * 2048;
            if (idx + 1 < 64) {
                stage_tile_async(Kt + (size_t)(jt + 1) * 16 * 128, nxtb, lane);
                asy_wait<8>();      // current tile's 8 issues complete
            } else {
                asy_wait<0>();
            }
            v8f acc = {};
#pragma unroll
            for (int kk = 0; kk < 4; ++kk) {
                v16h bf = ldB_nt(curb + kk * 32, 128, lane);
                acc = wmma32(qa[kk], bf, acc);
            }
            const int n = lane & 15;
#pragma unroll
            for (int v = 0; v < 8; ++v) {
                int m = v + (alo ? 0 : 8);
                Sh[(size_t)m * 4096 + j0 + n] = (f16)(acc[v] * SCALE);
            }
        }
    }
#else
    for (int jt = wave * 64; jt < wave * 64 + 64; ++jt) {
        const int j0 = jt * 16;
        __builtin_prefetch(Kt + (size_t)(j0 + 16) * 128, 0, 3);
        v8f acc = {};
#pragma unroll
        for (int kk = 0; kk < 4; ++kk) {
            v16h bf = ldB_nt(Kt + (size_t)j0 * 128 + kk * 32, 128, lane);
            acc = wmma32(qa[kk], bf, acc);
        }
        const int n = lane & 15;
#pragma unroll
        for (int v = 0; v < 8; ++v) {
            int m = v + (alo ? 0 : 8);
            Sh[(size_t)m * 4096 + j0 + n] = (f16)(acc[v] * SCALE);
        }
    }
#endif
    __syncthreads();

    // ---- phase 2: softmax on f16 rows (max, exp in place, sums) ----
    {
        const int r = tid & 15, chunk = tid >> 4;     // 8 chunks of 512
        f16* row = Sh + (size_t)r * 4096 + chunk * 512;
        float mx = -1e30f;
        for (int q = 0; q < 512; ++q) mx = fmaxf(mx, (float)row[q]);
        s_part[r * 8 + chunk] = mx;
        __syncthreads();
        if (tid < 16) {
            float m2 = s_part[tid * 8];
#pragma unroll
            for (int c = 1; c < 8; ++c) m2 = fmaxf(m2, s_part[tid * 8 + c]);
            s_max[tid] = m2;
        }
        __syncthreads();
        const float rm = s_max[r];
        float sum = 0.0f;
        for (int q = 0; q < 512; ++q) {
            float e = __expf((float)row[q] - rm);
            row[q] = (f16)e;
            sum += e;
        }
        s_part[r * 8 + chunk] = sum;
        __syncthreads();
        if (tid < 16) {
            float t2 = 0.0f;
#pragma unroll
            for (int c = 0; c < 8; ++c) t2 += s_part[tid * 8 + c];
            s_inv[tid] = 1.0f / t2;
        }
        __syncthreads();
    }

    // ---- phase 3: O^T[i,c] = sum_j P[i,j] V2[c,j]; each wave: 2 c-tiles ----
    v8f acc0 = {}, acc1 = {};
    const int n0 = wave * 2;
    for (int j0 = 0; j0 < 4096; j0 += 32) {
        v16h pa = ldA(Sh + j0, 4096, lane);     // P tile from LDS (f16)
        v16h b0 = ldB_nt(V2 + (size_t)(n0 * 16) * 4096 + j0, 4096, lane);
        v16h b1 = ldB_nt(V2 + (size_t)((n0 + 1) * 16) * 4096 + j0, 4096, lane);
        acc0 = wmma32(pa, b0, acc0);
        acc1 = wmma32(pa, b1, acc1);
    }
#pragma unroll
    for (int v = 0; v < 8; ++v) {
        int m = v + (alo ? 0 : 8);
        float inv = s_inv[m];
        int i = i0 + m;
        int n = lane & 15;
        Ot[(size_t)i * 128 + n0 * 16 + n]       = (f16)(acc0[v] * inv);
        Ot[(size_t)i * 128 + (n0 + 1) * 16 + n] = (f16)(acc1[v] * inv);
    }
}

// ---------------------------------------------------------------------------
// Kernel 4: generic channels-last 3D conv as implicit GEMM (Cout = 64).
// A rows = output voxels (im2col on the fly), B rows = repacked weights.
// Fused epilogue: +bias, BN1, relu, BN2, relu.  FINAL -> fp32 NCDHW out.
// ---------------------------------------------------------------------------
template <bool FINAL>
__global__ __launch_bounds__(128) void conv3d_cl(
    const f16* __restrict__ in, const f16* __restrict__ wgt,
    const float* __restrict__ bias,
    const float* __restrict__ g1, const float* __restrict__ b1,
    const float* __restrict__ g2, const float* __restrict__ b2,
    f16* __restrict__ out, float* __restrict__ outF,
    int Cin, int iD0, int iD1, int iD2, int oD0, int oD1, int oD2,
    int s0, int s1, int s2, int p0, int p1, int p2)
{
    const int lane = threadIdx.x & 31;
    const int wave = threadIdx.x >> 5;        // n-tile: co base = wave*16
    const int alo  = (lane < 16);
    const int b    = blockIdx.y;
    const int m0   = blockIdx.x * 16;
    const int Ksteps = (27 * Cin) >> 5;
    const int ldw = 27 * Cin;

    // per-lane output coordinates (row = lane&15)
    const int m = m0 + (lane & 15);
    const int z = m / (oD1 * oD2);
    const int rr0 = m % (oD1 * oD2);
    const int y = rr0 / oD2, x = rr0 % oD2;
    const int zb = z * s0 - p0, yb = y * s1 - p1, xb = x * s2 - p2;

    v8f acc = {};
    for (int kb = 0; kb < Ksteps; ++kb) {
        const int kbase = kb << 5;
        const int koff = kbase / Cin;
        const int ci0 = kbase - koff * Cin;
        const int kd = koff / 9, kh = (koff / 3) % 3, kw = koff % 3;
        const int zi = zb + kd, yi = yb + kh, xi = xb + kw;

        v16h a = {};
        if (zi >= 0 && zi < iD0 && yi >= 0 && yi < iD1 && xi >= 0 && xi < iD2) {
            const f16* p = in + ((((size_t)b * iD0 + zi) * iD1 + yi) * iD2 + xi)
                               * Cin + ci0 + (alo ? 0 : 8);
            v8h lo = *(const v8h*)(p);
            v8h hi = *(const v8h*)(p + 16);
            a = __builtin_shufflevector(lo, hi, 0,1,2,3,4,5,6,7,8,9,10,11,12,13,14,15);
        }
        v16h bf = ldB_nt(wgt + (size_t)(wave * 16) * ldw + kbase, ldw, lane);
        acc = wmma32(a, bf, acc);
    }

    const int co = wave * 16 + (lane & 15);
    const float bb = bias[co];
    const float sc1 = g1[co] * RSBN, sh1 = b1[co];
    const float sc2 = g2[co] * RSBN, sh2 = b2[co];
#pragma unroll
    for (int v = 0; v < 8; ++v) {
        int mm = m0 + v + (alo ? 0 : 8);
        int zz = mm / (oD1 * oD2);
        int rr = mm % (oD1 * oD2);
        int yy = rr / oD2, xx = rr % oD2;
        float val = acc[v] + bb;
        val = fmaxf(val * sc1 + sh1, 0.0f);
        val = fmaxf(val * sc2 + sh2, 0.0f);
        if constexpr (FINAL)
            outF[(((size_t)(b * 64 + co) * oD0 + zz) * oD1 + yy) * oD2 + xx] = val;
        else
            out[((((size_t)b * oD0 + zz) * oD1 + yy) * oD2 + xx) * 64 + co] = (f16)val;
    }
}

// ---------------------------------------------------------------------------
// Small utility kernels
// ---------------------------------------------------------------------------
__global__ void cvt_f16(const float* __restrict__ s, f16* __restrict__ d, int n) {
    int i = blockIdx.x * 256 + threadIdx.x;
    if (i < n) d[i] = (f16)s[i];
}

// [O, I, 3,3,3] f32  ->  [O, 27, I] f16  (so K-chunks of 32 channels are contiguous)
__global__ void repack_w(const float* __restrict__ s, f16* __restrict__ d,
                         int Cout, int Cin) {
    int i = blockIdx.x * 256 + threadIdx.x;
    int tot = Cout * 27 * Cin;
    if (i < tot) {
        int co = i / (27 * Cin);
        int rem = i - co * 27 * Cin;
        int koff = rem / Cin;
        int ci = rem - koff * Cin;
        d[i] = (f16)s[((size_t)(co * Cin + ci)) * 27 + koff];
    }
}

__global__ void zero_u32(uint32_t* __restrict__ p, int n) {
    int i = blockIdx.x * 256 + threadIdx.x;
    if (i < n) p[i] = 0u;
}

__global__ __launch_bounds__(128) void scatter_k(
    const int* __restrict__ coords, const f16* __restrict__ sp,
    f16* __restrict__ dense)
{
    const int bk = blockIdx.x;
    const int b = bk >> 12;
    const int* c = coords + (size_t)bk * 3;
    const int d = c[0], w = c[1], hh = c[2];
    const size_t dst = ((((size_t)b * 48 + hh) * 48 + d) * 64 + w) * 128;
    dense[dst + threadIdx.x] = sp[(size_t)bk * 128 + threadIdx.x];
}

// ---------------------------------------------------------------------------
// Host-side orchestration
// ---------------------------------------------------------------------------
extern "C" void kernel_launch(void* const* d_in, const int* in_sizes, int n_in,
                              void* d_out, int out_size, void* d_ws, size_t ws_size,
                              hipStream_t stream) {
    (void)in_sizes; (void)n_in; (void)out_size; (void)ws_size;
    const float* vox = (const float*)d_in[0];
    const int* coords = (const int*)d_in[1];
    auto F = [&](int i) { return (const float*)d_in[i]; };

    // workspace carve-out
    char* wsb = (char*)d_ws;
    size_t off = 0;
    auto alloc = [&](size_t bytes) -> void* {
        void* p = wsb + off;
        off = (off + bytes + 255) & ~(size_t)255;
        return p;
    };
    const size_t xbytes = (size_t)2 * 4096 * 128 * sizeof(f16);
    f16* X0 = (f16*)alloc(xbytes);
    f16* XA = (f16*)alloc(xbytes);
    f16* XB = (f16*)alloc(xbytes);
    f16* Qt = (f16*)alloc(xbytes);
    f16* Kt = (f16*)alloc(xbytes);
    f16* V2 = (f16*)alloc(xbytes);
    f16* Ot = (f16*)alloc(xbytes);
    f16* Watt = (f16*)alloc((size_t)8 * 128 * 128 * sizeof(f16));
    f16* Wc0 = (f16*)alloc((size_t)64 * 27 * 128 * sizeof(f16));
    f16* Wc1 = (f16*)alloc((size_t)64 * 27 * 64 * sizeof(f16));
    f16* Wc2 = (f16*)alloc((size_t)64 * 27 * 64 * sizeof(f16));
    f16* Wc3 = (f16*)alloc((size_t)64 * 27 * 64 * sizeof(f16));
    f16* dense = (f16*)alloc((size_t)2 * 48 * 48 * 64 * 128 * sizeof(f16));
    f16* c1 = (f16*)alloc((size_t)2 * 16 * 48 * 64 * 64 * sizeof(f16));
    f16* c2 = (f16*)alloc((size_t)2 * 7 * 48 * 64 * 64 * sizeof(f16));
    f16* c3 = (f16*)alloc((size_t)2 * 3 * 48 * 64 * 64 * sizeof(f16));

    // ---- stage 1: fused VFE/FCN/LN ----
    // pytree (alphabetical) order: vfe1/vfe2/fcn each {W,b,beta,g}
    vfe_fused<<<dim3(2 * 4096), 128, 0, stream>>>(
        vox,
        F(48), F(49), F(51), F(50),   // vfe1: W,b,g,beta
        F(52), F(53), F(55), F(54),   // vfe2
        F(42), F(43), F(45), F(44),   // fcn
        F(47), F(46),                 // ln_g, ln_b
        X0);

    // ---- weight conversion / repack ----
    for (int i = 0; i < 4; ++i)   // att1: Wk,Wo,Wq,Wv at d_in[2..5]
        cvt_f16<<<(16384 + 255) / 256, 256, 0, stream>>>(F(2 + i), Watt + (size_t)i * 16384, 16384);
    for (int i = 0; i < 4; ++i)   // att2 at d_in[10..13]
        cvt_f16<<<(16384 + 255) / 256, 256, 0, stream>>>(F(10 + i), Watt + (size_t)(4 + i) * 16384, 16384);
    repack_w<<<(64 * 27 * 128 + 255) / 256, 256, 0, stream>>>(F(18 + 5), Wc0, 64, 128);
    repack_w<<<(64 * 27 * 64 + 255) / 256, 256, 0, stream>>>(F(24 + 5), Wc1, 64, 64);
    repack_w<<<(64 * 27 * 64 + 255) / 256, 256, 0, stream>>>(F(30 + 5), Wc2, 64, 64);
    repack_w<<<(64 * 27 * 64 + 255) / 256, 256, 0, stream>>>(F(36 + 5), Wc3, 64, 64);

    // ---- stage 2: two attention blocks ----
    // LDS: 16*4096 f16 (S) + 160 f32 stats + 4*2*2048 f16 staging
    const size_t SMEM_ATTN = (size_t)16 * 4096 * 2 + 160 * 4 + (size_t)4 * 2 * 2048 * 2;
    f16* cur = X0;
    f16* nxt = XA;
    for (int blk = 0; blk < 2; ++blk) {
        const f16* Wk16 = Watt + (size_t)(blk * 4 + 0) * 16384;
        const f16* Wo16 = Watt + (size_t)(blk * 4 + 1) * 16384;
        const f16* Wq16 = Watt + (size_t)(blk * 4 + 2) * 16384;
        const f16* Wv16 = Watt + (size_t)(blk * 4 + 3) * 16384;
        const float* bk = F(6 + 8 * blk);
        const float* bo = F(7 + 8 * blk);
        const float* bq = F(8 + 8 * blk);
        const float* bv = F(9 + 8 * blk);
        for (int b = 0; b < 2; ++b) {
            const size_t xo = (size_t)b * 4096 * 128;
            gemm_nt_128<false, false><<<64, 128, 0, stream>>>(cur + xo, Wq16, bq, nullptr, Qt + xo, 4096);
            gemm_nt_128<false, false><<<64, 128, 0, stream>>>(cur + xo, Wk16, bk, nullptr, Kt + xo, 4096);
            gemm_nt_128<false, true ><<<64, 128, 0, stream>>>(cur + xo, Wv16, bv, nullptr, V2 + xo, 4096);
            attn_core<<<256, 128, SMEM_ATTN, stream>>>(Qt + xo, Kt + xo, V2 + xo, Ot + xo);
            gemm_nt_128<true, false><<<64, 128, 0, stream>>>(Ot + xo, Wo16, bo, cur + xo, nxt + xo, 4096);
        }
        cur = nxt;   // blk0: cur=XA ; blk1: cur=XB
        nxt = XB;
    }

    // ---- stage 3: scatter to dense (channels-last f16) ----
    const int denseU32 = (int)((size_t)2 * 48 * 48 * 64 * 128 * sizeof(f16) / 4);
    zero_u32<<<(denseU32 + 255) / 256, 256, 0, stream>>>((uint32_t*)dense, denseU32);
    scatter_k<<<dim3(2 * 4096), 128, 0, stream>>>(coords, cur, dense);

    // ---- stage 4: 4 conv stages (implicit GEMM, Cout=64) ----
    // stage 0: 128ch (48,48,64) -> (16,48,64), stride(3,1,1), pad(1,1,1)
    conv3d_cl<false><<<dim3(3072, 2), 128, 0, stream>>>(
        dense, Wc0, F(18), F(21), F(19), F(22), F(20), c1, nullptr,
        128, 48, 48, 64, 16, 48, 64, 3, 1, 1, 1, 1, 1);
    // stage 1: 64ch (16,48,64) -> (7,48,64), stride(2,1,1), pad(0,1,1)
    conv3d_cl<false><<<dim3(1344, 2), 128, 0, stream>>>(
        c1, Wc1, F(24), F(27), F(25), F(28), F(26), c2, nullptr,
        64, 16, 48, 64, 7, 48, 64, 2, 1, 1, 0, 1, 1);
    // stage 2: 64ch (7,48,64) -> (3,48,64), stride(2,1,1), pad(0,1,1)
    conv3d_cl<false><<<dim3(576, 2), 128, 0, stream>>>(
        c2, Wc2, F(30), F(33), F(31), F(34), F(32), c3, nullptr,
        64, 7, 48, 64, 3, 48, 64, 2, 1, 1, 0, 1, 1);
    // stage 3: 64ch (3,48,64) -> (1,48,64), stride(3,1,1), pad(0,1,1) -> fp32 out
    conv3d_cl<true><<<dim3(192, 2), 128, 0, stream>>>(
        c3, Wc3, F(36), F(39), F(37), F(40), F(38), nullptr, (float*)d_out,
        64, 3, 48, 64, 1, 48, 64, 3, 1, 1, 0, 1, 1);
}
